// QANet_9028021256204
// MI455X (gfx1250) — matrix operand order repaired and checked by reading, not compiled, not run on hardware
//
#include <hip/hip_runtime.h>
#include <hip/hip_bf16.h>
#include <math.h>

// =====================================================================
// QANet forward for MI455X (gfx1250, wave32, WMMA).
// All contractions go through one stride-generic WMMA GEMM kernel using
// v_wmma_f32_16x16x32_f16 (f32 accumulate) with LDS-staged f16 tiles.
// Requires ws_size >= ~200 MB.
// =====================================================================

typedef long long i64;
typedef __attribute__((ext_vector_type(16))) _Float16 v16h;
typedef __attribute__((ext_vector_type(8)))  float    v8f;

union U16 { unsigned u[8]; v16h h; };

static inline unsigned cdiv(i64 a, i64 b) { return (unsigned)((a + b - 1) / b); }

__device__ __forceinline__ unsigned pack2f16(float x, float y)
{
  _Float16 a = (_Float16)x, b = (_Float16)y;
  unsigned short ua = __builtin_bit_cast(unsigned short, a);
  unsigned short ub = __builtin_bit_cast(unsigned short, b);
  return (unsigned)ua | ((unsigned)ub << 16);
}

// ------------------------------- WMMA GEMM ---------------------------
// C[b1,b2,m,n] = act(alpha * sum_k A[..,m,k]*B[..,k,n] + bias[n]) + R[..]
// Block: 128 threads = 4 waves; block tile 64(M) x 64(N), K-step 32.
// Each wave computes a 16x64 strip: 1 A fragment x 4 B fragments -> 4 WMMA.
// A/B tiles staged in LDS as packed f16 pairs (one u32 = 2 consecutive K).
// Interior tiles with K-stride 1 stage via float2 (global_load_b64);
// edge tiles use a branch-free masked scalar path. All four B fragments
// are materialized before the WMMAs so one ds-wait covers the burst and
// the 4 v_wmma issue back-to-back.
// Lane layouts per CDNA5 ISA 7.12.2:
//   A 16x32 f16: m = lane&15; elem e -> k = ((e&8)<<1) + (lane>>4)*8 + (e&7)
//   B 32x16 f16: n = lane&15; elem e -> k = ((e&8)<<1) + (lane>>4)*8 + (e&7) (same as A)
//   C 16x16 f32: vgpr v -> m = v + 8*(lane>>4); n = lane&15
__global__ __launch_bounds__(128) void k_gemm(
    const float* __restrict__ A, i64 aB1, i64 aB2, i64 aM, i64 aK,
    const float* __restrict__ Bm, i64 bB1, i64 bB2, i64 bK, i64 bN,
    float* __restrict__ C, i64 cB1, i64 cB2, i64 cM, i64 cN,
    const float* __restrict__ bias, const float* __restrict__ R,
    float alpha, int act, int M, int N, int K, int b2n)
{
  __shared__ unsigned AsU[64 * 17];   // [m][k/2], row stride 17 words
  __shared__ unsigned BsU[64 * 17];   // [n][k/2] (B transposed), row stride 17

  const int tid  = threadIdx.x;
  const int lane = tid & 31;
  const int wave = tid >> 5;
  const int half = lane >> 4;
  const int l15  = lane & 15;
  const int bz = blockIdx.z;
  const int b1 = bz / b2n, b2 = bz % b2n;
  const float* Ab = A  + (i64)b1 * aB1 + (i64)b2 * aB2;
  const float* Bb = Bm + (i64)b1 * bB1 + (i64)b2 * bB2;
  float*       Cb = C  + (i64)b1 * cB1 + (i64)b2 * cB2;
  const float* Rb = R ? (R + (i64)b1 * cB1 + (i64)b2 * cB2) : (const float*)0;

  const int mBase = blockIdx.x * 64;
  const int n0    = blockIdx.y * 64;

  // fast-path eligibility: contiguous K, even row stride => 8B-aligned float2
  const bool fa = (aK == 1) && ((aM & 1) == 0);
  const bool fb = (bK == 1) && ((bN & 1) == 0);
  const bool fullM = (mBase + 64 <= M);
  const bool fullN = (n0    + 64 <= N);

  v8f acc[4];
#pragma unroll
  for (int t = 0; t < 4; t++) acc[t] = (v8f){0.f,0.f,0.f,0.f,0.f,0.f,0.f,0.f};

  for (int kb = 0; kb < K; kb += 32) {
    const bool fullK = (kb + 32 <= K);

    // ---- stage A tile (64 x 32) as packed f16 pairs ----
    if (fa && fullK && fullM) {               // uniform fast path: b64 loads
#pragma unroll
      for (int it = 0; it < 8; it++) {
        int w  = tid + it * 128;              // word id in [0, 1024)
        int m  = w >> 4;
        int kp = (w & 15) << 1;
        const float2 f = *(const float2*)(Ab + (i64)(mBase + m) * aM + (kb + kp));
        AsU[m * 17 + (kp >> 1)] = pack2f16(f.x, f.y);
      }
    } else {                                  // masked scalar path, branch-free
#pragma unroll
      for (int it = 0; it < 8; it++) {
        int w  = tid + it * 128;
        int m  = w >> 4;
        int kp = (w & 15) << 1;
        int mg = mBase + m, kg = kb + kp;
        bool p0 = (mg < M) && (kg     < K);
        bool p1 = (mg < M) && (kg + 1 < K);
        i64 o0 = p0 ? ((i64)mg * aM + (i64)kg * aK)       : 0;
        i64 o1 = p1 ? ((i64)mg * aM + (i64)(kg + 1) * aK) : 0;
        float f0 = Ab[o0]; f0 = p0 ? f0 : 0.f;
        float f1 = Ab[o1]; f1 = p1 ? f1 : 0.f;
        AsU[m * 17 + (kp >> 1)] = pack2f16(f0, f1);
      }
    }
    // ---- stage B tile (32 x 64) transposed to [n][k] ----
    if (fb && fullK && fullN) {
#pragma unroll
      for (int it = 0; it < 8; it++) {
        int w  = tid + it * 128;
        int n  = w >> 4;
        int kp = (w & 15) << 1;
        const float2 f = *(const float2*)(Bb + (i64)(n0 + n) * bN + (kb + kp));
        BsU[n * 17 + (kp >> 1)] = pack2f16(f.x, f.y);
      }
    } else {
#pragma unroll
      for (int it = 0; it < 8; it++) {
        int w  = tid + it * 128;
        int n  = w >> 4;
        int kp = (w & 15) << 1;
        int ng = n0 + n, kg = kb + kp;
        bool p0 = (ng < N) && (kg     < K);
        bool p1 = (ng < N) && (kg + 1 < K);
        i64 o0 = p0 ? ((i64)kg * bK + (i64)ng * bN)       : 0;
        i64 o1 = p1 ? ((i64)(kg + 1) * bK + (i64)ng * bN) : 0;
        float f0 = Bb[o0]; f0 = p0 ? f0 : 0.f;
        float f1 = Bb[o1]; f1 = p1 ? f1 : 0.f;
        BsU[n * 17 + (kp >> 1)] = pack2f16(f0, f1);
      }
    }
    __syncthreads();

    // ---- A fragment: 8 x ds_load_b32 ----
    U16 af;
    const int ml = wave * 16 + l15;
#pragma unroll
    for (int v = 0; v < 8; v++) {
      int kw = ((v & 4) << 1) + half * 4 + (v & 3);   // k/2 per ISA A layout
      af.u[v] = AsU[ml * 17 + kw];
    }
    // ---- load ALL 4 B fragments, then 4 back-to-back WMMAs ----
    U16 bf[4];
#pragma unroll
    for (int t = 0; t < 4; t++) {
      const int nl = t * 16 + l15;
#pragma unroll
      for (int v = 0; v < 8; v++)
        bf[t].u[v] = BsU[nl * 17 + ((v & 4) << 1) + half * 4 + (v & 3)];      // k/2 per ISA A layout (matches A)
    }
#pragma unroll
    for (int t = 0; t < 4; t++)
      acc[t] = __builtin_amdgcn_wmma_f32_16x16x32_f16(
          false, af.h, false, bf[t].h, (short)0, acc[t], false, false);
    __syncthreads();
  }

  // ---- epilogue: bias / alpha / activation / residual ----
  const int m0w = mBase + wave * 16;
#pragma unroll
  for (int v = 0; v < 8; v++) {
    int m = m0w + v + (half << 3);
    if (m >= M) continue;
#pragma unroll
    for (int t = 0; t < 4; t++) {
      int n = n0 + l15 + t * 16;
      if (n >= N) continue;
      float r = acc[t][v] * alpha;
      if (bias) r += bias[n];
      if (act == 1)      r = fmaxf(r, 0.f);
      else if (act == 2) r = 1.f / (1.f + __expf(-r));
      if (Rb) r += Rb[(i64)m * cM + (i64)n * cN];
      Cb[(i64)m * cM + (i64)n * cN] = r;
    }
  }
}

// --------------------------- elementwise kernels ---------------------
__global__ void k_embed(const float* __restrict__ wm, const float* __restrict__ chm,
                        const int* __restrict__ widx, const int* __restrict__ cidx,
                        float* __restrict__ out, int Bn, int L, int WLn, int WEn, int CEn)
{
  const int DINn = WEn + CEn;
  i64 idx = (i64)blockIdx.x * blockDim.x + threadIdx.x;
  i64 total = (i64)Bn * L * DINn;
  if (idx >= total) return;
  int f = (int)(idx % DINn);
  i64 bl = idx / DINn;
  if (f < WEn) {
    out[idx] = wm[(i64)widx[bl] * WEn + f];
  } else {
    int cf = f - WEn;
    float mx = -3.4e38f;
    for (int t = 0; t < WLn; t++)
      mx = fmaxf(mx, chm[(i64)cidx[bl * WLn + t] * CEn + cf]);
    out[idx] = mx;
  }
}

__global__ void k_hwcomb(float* __restrict__ x, const float* __restrict__ g,
                         const float* __restrict__ nn, i64 total)
{
  i64 idx = (i64)blockIdx.x * blockDim.x + threadIdx.x;
  if (idx >= total) return;
  float gv = g[idx];
  x[idx] = gv * nn[idx] + (1.f - gv) * x[idx];
}

__global__ void k_posenc(const float* __restrict__ x, float* __restrict__ y,
                         int Bn, int L, int Cc)
{
  i64 idx = (i64)blockIdx.x * blockDim.x + threadIdx.x;
  i64 total = (i64)Bn * L * Cc;
  if (idx >= total) return;
  int c = (int)(idx % Cc);
  int l = (int)((idx / Cc) % L);
  int jj = c & ~1;
  float ang = (float)l * __expf(-logf(10000.f) * (float)jj / (float)Cc);
  float pe = (c & 1) ? __cosf(ang) : __sinf(ang);
  y[idx] = x[idx] + pe;
}

__global__ void k_dwconv(const float* __restrict__ x, float* __restrict__ y,
                         const float* __restrict__ w, const float* __restrict__ bv,
                         int Bn, int L, int Cc, int ksz)
{
  i64 idx = (i64)blockIdx.x * blockDim.x + threadIdx.x;
  i64 total = (i64)Bn * L * Cc;
  if (idx >= total) return;
  int c = (int)(idx % Cc);
  int l = (int)((idx / Cc) % L);
  i64 bb = idx / ((i64)Cc * L);
  int pad = ksz >> 1;
  float acc = bv[c];
  for (int t = 0; t < ksz; t++) {
    int ll = l + t - pad;
    if (ll >= 0 && ll < L) acc += x[(bb * L + ll) * Cc + c] * w[c * ksz + t];
  }
  y[idx] = acc;
}

// layernorm over entire (L*C) slab per batch; gmode: 0 -> g index = l*C+c, 1 -> c*L+l
__global__ void k_ln2(const float* __restrict__ x, float* __restrict__ y,
                      const float* __restrict__ g, const float* __restrict__ b,
                      int L, int Cc, int gmode)
{
  const int bb = blockIdx.x;
  const i64 n = (i64)L * Cc;
  const float* xb = x + (i64)bb * n;
  float* yb = y + (i64)bb * n;
  float s = 0.f, s2 = 0.f;
  for (i64 i = threadIdx.x; i < n; i += blockDim.x) { float v = xb[i]; s += v; s2 += v * v; }
  __shared__ float sh1[256], sh2[256];
  sh1[threadIdx.x] = s; sh2[threadIdx.x] = s2; __syncthreads();
  for (int o = blockDim.x >> 1; o > 0; o >>= 1) {
    if ((int)threadIdx.x < o) { sh1[threadIdx.x] += sh1[threadIdx.x + o]; sh2[threadIdx.x] += sh2[threadIdx.x + o]; }
    __syncthreads();
  }
  float mean = sh1[0] / (float)n;
  float var  = sh2[0] / (float)n - mean * mean;
  float inv = rsqrtf(var + 1e-5f);
  for (i64 i = threadIdx.x; i < n; i += blockDim.x) {
    int l = (int)(i / Cc), c = (int)(i % Cc);
    i64 gi = gmode ? ((i64)c * L + l) : i;
    yb[i] = (xb[i] - mean) * inv * g[gi] + b[gi];
  }
}

// strided softmax: slice s covers elements base + i*stride, i in [0,n),
// base = (s/inner)*ostride + (s%inner). out may alias in.
__global__ void k_softmax(const float* __restrict__ in, float* __restrict__ out,
                          int n, i64 stride, i64 ostride, int inner, int nslices)
{
  int s = blockIdx.x;
  if (s >= nslices) return;
  i64 base = (i64)(s / inner) * ostride + (i64)(s % inner);
  __shared__ float sh[256];
  float mx = -3.4e38f;
  for (int i = threadIdx.x; i < n; i += blockDim.x)
    mx = fmaxf(mx, in[base + (i64)i * stride]);
  sh[threadIdx.x] = mx; __syncthreads();
  for (int o = blockDim.x >> 1; o > 0; o >>= 1) {
    if ((int)threadIdx.x < o) sh[threadIdx.x] = fmaxf(sh[threadIdx.x], sh[threadIdx.x + o]);
    __syncthreads();
  }
  mx = sh[0]; __syncthreads();
  float sm = 0.f;
  for (int i = threadIdx.x; i < n; i += blockDim.x)
    sm += __expf(in[base + (i64)i * stride] - mx);
  sh[threadIdx.x] = sm; __syncthreads();
  for (int o = blockDim.x >> 1; o > 0; o >>= 1) {
    if ((int)threadIdx.x < o) sh[threadIdx.x] += sh[threadIdx.x + o];
    __syncthreads();
  }
  float inv = 1.f / sh[0];
  for (int i = threadIdx.x; i < n; i += blockDim.x) {
    i64 p = base + (i64)i * stride;
    out[p] = __expf(in[p] - mx) * inv;
  }
}

__global__ void k_logsoftmax(const float* __restrict__ in, float* __restrict__ out, int n)
{
  int r = blockIdx.x;
  const float* xr = in + (i64)r * n;
  float* yr = out + (i64)r * n;
  __shared__ float sh[256];
  float mx = -3.4e38f;
  for (int i = threadIdx.x; i < n; i += blockDim.x) mx = fmaxf(mx, xr[i]);
  sh[threadIdx.x] = mx; __syncthreads();
  for (int o = blockDim.x >> 1; o > 0; o >>= 1) {
    if ((int)threadIdx.x < o) sh[threadIdx.x] = fmaxf(sh[threadIdx.x], sh[threadIdx.x + o]);
    __syncthreads();
  }
  mx = sh[0]; __syncthreads();
  float sm = 0.f;
  for (int i = threadIdx.x; i < n; i += blockDim.x) sm += __expf(xr[i] - mx);
  sh[threadIdx.x] = sm; __syncthreads();
  for (int o = blockDim.x >> 1; o > 0; o >>= 1) {
    if ((int)threadIdx.x < o) sh[threadIdx.x] += sh[threadIdx.x + o];
    __syncthreads();
  }
  float lse = mx + logf(sh[0]);
  for (int i = threadIdx.x; i < n; i += blockDim.x) yr[i] = xr[i] - lse;
}

__global__ void k_matvec(const float* __restrict__ x, const float* __restrict__ w,
                         float* __restrict__ out, int rows, int K)
{
  int r = blockIdx.x * blockDim.x + threadIdx.x;
  if (r >= rows) return;
  float a = 0.f;
  for (int k = 0; k < K; k++) a += x[(i64)r * K + k] * w[k];
  out[r] = a;
}

__global__ void k_scalecols(const float* __restrict__ x, const float* __restrict__ w,
                            float* __restrict__ y, i64 rows, int K)
{
  i64 idx = (i64)blockIdx.x * blockDim.x + threadIdx.x;
  if (idx >= rows * K) return;
  y[idx] = x[idx] * w[idx % K];
}

__global__ void k_cq_add(float* __restrict__ S, const float* __restrict__ cwc,
                         const float* __restrict__ qwq, int Bn, int Lc, int Lq)
{
  i64 idx = (i64)blockIdx.x * blockDim.x + threadIdx.x;
  i64 total = (i64)Bn * Lc * Lq;
  if (idx >= total) return;
  int j = (int)(idx % Lq);
  int i = (int)((idx / Lq) % Lc);
  int b = (int)(idx / ((i64)Lq * Lc));
  S[idx] += cwc[b * Lc + i] + qwq[b * Lq + j];
}

__global__ void k_cqcat(const float* __restrict__ c, const float* __restrict__ A,
                        const float* __restrict__ Bt, float* __restrict__ out,
                        int Bn, int L, int Dd)
{
  i64 idx = (i64)blockIdx.x * blockDim.x + threadIdx.x;
  i64 total = (i64)Bn * L * Dd;
  if (idx >= total) return;
  int dd = (int)(idx % Dd);
  i64 bl = idx / Dd;
  float cv = c[idx], av = A[idx], bv = Bt[idx];
  i64 o = bl * (i64)(4 * Dd);
  out[o + dd] = cv;
  out[o + Dd + dd] = av;
  out[o + 2 * Dd + dd] = cv * av;
  out[o + 3 * Dd + dd] = cv * bv;
}

__global__ void k_catMM(const float* __restrict__ M1, const float* __restrict__ M2,
                        float* __restrict__ out, int Bn, int L, int Dd)
{
  i64 idx = (i64)blockIdx.x * blockDim.x + threadIdx.x;
  i64 total = (i64)Bn * L * Dd;
  if (idx >= total) return;
  int dd = (int)(idx % Dd);
  i64 bl = idx / Dd;
  out[bl * (i64)(2 * Dd) + dd] = M1[idx];
  out[bl * (i64)(2 * Dd) + Dd + dd] = M2[idx];
}

// ------------------------------ host side ----------------------------
struct ConvP { const float *dw_w, *dw_b, *pw_w, *pw_b; };
struct BlockP {
  const float *fc_dw_w, *fc_dw_b, *fc_pw_w, *fc_pw_b;
  ConvP conv[4];
  const float *cn_g, *cn_b, *ln_g, *ln_b, *wq, *wk, *wv, *wo, *fc_w, *fc_b;
  int nconv, ksz;
};
struct Scratch { float *g1, *g2, *g3, *Qb, *Kb, *Vb, *OB, *SC; };

static void gemm(hipStream_t st, int M, int N, int K, int b1n, int b2n,
                 const float* A, i64 a1, i64 a2, i64 am, i64 ak,
                 const float* Bp, i64 bb1, i64 bb2, i64 bk, i64 bn,
                 float* C, i64 c1, i64 c2, i64 cm, i64 cn,
                 const float* bias, const float* R, float alpha, int act)
{
  dim3 g(cdiv(M, 64), cdiv(N, 64), (unsigned)(b1n * b2n));
  k_gemm<<<g, dim3(128), 0, st>>>(A, a1, a2, am, ak, Bp, bb1, bb2, bk, bn,
                                  C, c1, c2, cm, cn, bias, R, alpha, act, M, N, K, b2n);
}

#define EW(total) dim3(cdiv((total), 256)), dim3(256), 0, st

static void run_block(hipStream_t st, const Scratch& s, const BlockP& p,
                      const float* xin, float* xout, int Bn, int L, int din, int d)
{
  const int H = 8, DK = 16;
  const i64 nin = (i64)Bn * L * din;
  const i64 nd  = (i64)Bn * L * d;

  // x + positional encoding -> g3 ; depthwise -> g2 ; pointwise din->d -> g1
  k_posenc<<<EW(nin)>>>(xin, s.g3, Bn, L, din);
  k_dwconv<<<EW(nin)>>>(s.g3, s.g2, p.fc_dw_w, p.fc_dw_b, Bn, L, din, p.ksz);
  gemm(st, Bn * L, d, din, 1, 1,
       s.g2, 0, 0, din, 1,
       p.fc_pw_w, 0, 0, 1, din,
       s.g1, 0, 0, d, 1,
       p.fc_pw_b, nullptr, 1.f, 0);

  float *X = s.g1, *u = s.g2, *w2 = s.g3;
  for (int c = 0; c < p.nconv; c++) {
    k_ln2<<<dim3(Bn), dim3(256), 0, st>>>(X, u, p.cn_g, p.cn_b, L, d, 1);
    k_dwconv<<<EW(nd)>>>(u, w2, p.conv[c].dw_w, p.conv[c].dw_b, Bn, L, d, p.ksz);
    gemm(st, Bn * L, d, d, 1, 1,
         w2, 0, 0, d, 1,
         p.conv[c].pw_w, 0, 0, 1, d,
         u, 0, 0, d, 1,
         p.conv[c].pw_b, X, 1.f, 1);        // x = res + relu(pw + b)
    float* tmp = X; X = u; u = tmp;
  }

  // --- self-attention ---
  k_ln2<<<dim3(Bn), dim3(256), 0, st>>>(X, u, p.ln_g, p.ln_b, L, d, 0);
  const float* Wqkv[3] = { p.wq, p.wk, p.wv };
  float* QKV[3] = { s.Qb, s.Kb, s.Vb };
  for (int t = 0; t < 3; t++)
    gemm(st, L, DK, d, Bn, H,
         u, (i64)L * d, 0, d, 1,
         Wqkv[t], 0, (i64)d * DK, DK, 1,
         QKV[t], (i64)H * L * DK, (i64)L * DK, DK, 1,
         nullptr, nullptr, 1.f, 0);
  // scores s[b,h,i,j] = (q . k) / sqrt(dk)
  gemm(st, L, L, DK, Bn, H,
       s.Qb, (i64)H * L * DK, (i64)L * DK, DK, 1,
       s.Kb, (i64)H * L * DK, (i64)L * DK, 1, DK,
       s.SC, (i64)H * L * L, (i64)L * L, L, 1,
       nullptr, nullptr, 1.f / sqrtf((float)DK), 0);
  // softmax over axis=2 (i), per reference quirk: slices (b,h,j)
  k_softmax<<<dim3((unsigned)(Bn * H * L)), dim3(256), 0, st>>>(
      s.SC, s.SC, L, (i64)L, (i64)L * L, L, Bn * H * L);
  // o = a @ v, heads interleaved into (B,L,D)
  gemm(st, L, DK, L, Bn, H,
       s.SC, (i64)H * L * L, (i64)L * L, L, 1,
       s.Vb, (i64)H * L * DK, (i64)L * DK, DK, 1,
       s.OB, (i64)L * d, DK, d, 1,
       nullptr, nullptr, 1.f, 0);
  // out proj + residual
  gemm(st, Bn * L, d, d, 1, 1,
       s.OB, 0, 0, d, 1,
       p.wo, 0, 0, d, 1,
       u, 0, 0, d, 1,
       nullptr, X, 1.f, 0);
  { float* tmp = X; X = u; u = tmp; }

  // --- feed-forward ---
  k_ln2<<<dim3(Bn), dim3(256), 0, st>>>(X, u, p.ln_g, p.ln_b, L, d, 0);
  gemm(st, Bn * L, d, d, 1, 1,
       u, 0, 0, d, 1,
       p.fc_w, 0, 0, 1, d,
       xout, 0, 0, d, 1,
       p.fc_b, X, 1.f, 1);
}

extern "C" void kernel_launch(void* const* d_in, const int* in_sizes, int n_in,
                              void* d_out, int out_size, void* d_ws, size_t ws_size,
                              hipStream_t stream)
{
  (void)in_sizes; (void)n_in; (void)out_size; (void)ws_size;
  hipStream_t st = stream;
  const int B = 16, Lc = 400, Lq = 50, WL = 16, WE = 300, CE = 64;
  const int DIN = WE + CE, D = 128;

  // ----- unpack inputs (setup_inputs dict insertion order, recursive) -----
  const int* cw_idx = (const int*)d_in[0];
  const int* cc_idx = (const int*)d_in[1];
  const int* qw_idx = (const int*)d_in[2];
  const int* qc_idx = (const int*)d_in[3];
  int cur = 4;
  auto nx = [&]() -> const float* { return (const float*)d_in[cur++]; };

  const float* word_mat = nx();
  const float* char_mat = nx();
  struct HWL { const float *lw, *lb, *gw, *gb; };
  HWL c_hw[2], q_hw[2];
  for (int i = 0; i < 2; i++) { c_hw[i].lw = nx(); c_hw[i].lb = nx(); c_hw[i].gw = nx(); c_hw[i].gb = nx(); }
  for (int i = 0; i < 2; i++) { q_hw[i].lw = nx(); q_hw[i].lb = nx(); q_hw[i].gw = nx(); q_hw[i].gb = nx(); }
  auto parse_block = [&](int nconv, int ksz) -> BlockP {
    BlockP p{}; p.nconv = nconv; p.ksz = ksz;
    p.fc_dw_w = nx(); p.fc_dw_b = nx(); p.fc_pw_w = nx(); p.fc_pw_b = nx();
    for (int c = 0; c < nconv; c++) { p.conv[c].dw_w = nx(); p.conv[c].dw_b = nx(); p.conv[c].pw_w = nx(); p.conv[c].pw_b = nx(); }
    p.cn_g = nx(); p.cn_b = nx(); p.ln_g = nx(); p.ln_b = nx();
    p.wq = nx(); p.wk = nx(); p.wv = nx(); p.wo = nx();
    p.fc_w = nx(); p.fc_b = nx();
    return p;
  };
  BlockP c_enc = parse_block(4, 7);
  BlockP q_enc = parse_block(4, 7);
  const float* W0 = nx();
  const float* resizer_w = nx();
  const float* resizer_b = nx();
  BlockP m0[7];
  for (int i = 0; i < 7; i++) m0[i] = parse_block(2, 5);
  const float* start_w = nx(); const float* start_b = nx();
  const float* end_w   = nx(); const float* end_b   = nx();

  // ----- workspace carve-up (floats) -----
  float* base = (float*)d_ws;
  size_t off = 0;
  auto alloc = [&](i64 n) -> float* { float* p = base + off; off += (size_t)n; return p; };
  float* embC = alloc((i64)B * Lc * DIN);
  float* embQ = alloc((i64)B * Lq * DIN);
  float* g1   = alloc((i64)B * Lc * DIN);
  float* g2   = alloc((i64)B * Lc * DIN);
  float* g3   = alloc((i64)B * Lc * DIN);
  float* zA   = alloc((i64)B * Lc * D);
  float* zB   = alloc((i64)B * Lc * D);
  float* M1   = alloc((i64)B * Lc * D);
  float* M2   = alloc((i64)B * Lc * D);
  float* M3   = alloc((i64)B * Lc * D);
  float* Cenc = alloc((i64)B * Lc * D);
  float* Qenc = alloc((i64)B * Lq * D);
  float* X0   = alloc((i64)B * Lc * D);
  float* Qb   = alloc((i64)B * Lc * D);
  float* Kb   = alloc((i64)B * Lc * D);
  float* Vb   = alloc((i64)B * Lc * D);
  float* OB   = alloc((i64)B * Lc * D);
  float* SC   = alloc((i64)B * 8 * Lc * Lc);   // 20.48M floats (attention scores)
  float* Ssim = alloc((i64)B * Lc * Lq);
  float* S2b  = alloc((i64)B * Lc * Lq);
  float* cwcB = alloc((i64)B * Lc);
  float* qwqB = alloc((i64)B * Lq);
  float* cmB  = alloc((i64)B * Lc * D);
  float* Aatt = alloc((i64)B * Lc * D);
  float* Btb  = alloc((i64)B * Lc * D);
  float* Tb   = alloc((i64)B * Lc * Lc);
  float* xcat = alloc((i64)B * Lc * 4 * D);
  float* hcat = alloc((i64)B * Lc * 2 * D);
  float* htmp = alloc((i64)B * Lc);

  Scratch sc{ g1, g2, g3, Qb, Kb, Vb, OB, SC };

  // ----- embeddings + highway -----
  k_embed<<<EW((i64)B * Lc * DIN)>>>(word_mat, char_mat, cw_idx, cc_idx, embC, B, Lc, WL, WE, CE);
  k_embed<<<EW((i64)B * Lq * DIN)>>>(word_mat, char_mat, qw_idx, qc_idx, embQ, B, Lq, WL, WE, CE);
  for (int l = 0; l < 2; l++) {
    gemm(st, B * Lc, DIN, DIN, 1, 1, embC, 0, 0, DIN, 1, c_hw[l].gw, 0, 0, 1, DIN,
         g1, 0, 0, DIN, 1, c_hw[l].gb, nullptr, 1.f, 2);
    gemm(st, B * Lc, DIN, DIN, 1, 1, embC, 0, 0, DIN, 1, c_hw[l].lw, 0, 0, 1, DIN,
         g2, 0, 0, DIN, 1, c_hw[l].lb, nullptr, 1.f, 1);
    k_hwcomb<<<EW((i64)B * Lc * DIN)>>>(embC, g1, g2, (i64)B * Lc * DIN);
  }
  for (int l = 0; l < 2; l++) {
    gemm(st, B * Lq, DIN, DIN, 1, 1, embQ, 0, 0, DIN, 1, q_hw[l].gw, 0, 0, 1, DIN,
         g1, 0, 0, DIN, 1, q_hw[l].gb, nullptr, 1.f, 2);
    gemm(st, B * Lq, DIN, DIN, 1, 1, embQ, 0, 0, DIN, 1, q_hw[l].lw, 0, 0, 1, DIN,
         g2, 0, 0, DIN, 1, q_hw[l].lb, nullptr, 1.f, 1);
    k_hwcomb<<<EW((i64)B * Lq * DIN)>>>(embQ, g1, g2, (i64)B * Lq * DIN);
  }

  // ----- embedding encoders -----
  run_block(st, sc, c_enc, embC, Cenc, B, Lc, DIN, D);
  run_block(st, sc, q_enc, embQ, Qenc, B, Lq, DIN, D);

  // ----- context-query attention -----
  k_matvec<<<dim3(cdiv(B * Lc, 256)), dim3(256), 0, st>>>(Cenc, W0,        cwcB, B * Lc, D);
  k_matvec<<<dim3(cdiv(B * Lq, 256)), dim3(256), 0, st>>>(Qenc, W0 + D,    qwqB, B * Lq, D);
  k_scalecols<<<EW((i64)B * Lc * D)>>>(Cenc, W0 + 2 * D, cmB, (i64)B * Lc, D);
  // S[b,i,j] = (c*wm) . q  (+ broadcast terms)
  gemm(st, Lc, Lq, D, B, 1,
       cmB, (i64)Lc * D, 0, D, 1,
       Qenc, (i64)Lq * D, 0, 1, D,
       Ssim, (i64)Lc * Lq, 0, Lq, 1,
       nullptr, nullptr, 1.f, 0);
  k_cq_add<<<EW((i64)B * Lc * Lq)>>>(Ssim, cwcB, qwqB, B, Lc, Lq);
  // S2 = softmax over i (axis 1) -> separate buffer; S1 = softmax over j in place
  k_softmax<<<dim3((unsigned)(B * Lq)), dim3(256), 0, st>>>(Ssim, S2b, Lc, (i64)Lq, (i64)Lc * Lq, Lq, B * Lq);
  k_softmax<<<dim3((unsigned)(B * Lc)), dim3(256), 0, st>>>(Ssim, Ssim, Lq, 1, (i64)Lq, 1, B * Lc);
  // A = S1 @ q
  gemm(st, Lc, D, Lq, B, 1,
       Ssim, (i64)Lc * Lq, 0, Lq, 1,
       Qenc, (i64)Lq * D, 0, D, 1,
       Aatt, (i64)Lc * D, 0, D, 1,
       nullptr, nullptr, 1.f, 0);
  // T = S1 @ S2^T
  gemm(st, Lc, Lc, Lq, B, 1,
       Ssim, (i64)Lc * Lq, 0, Lq, 1,
       S2b,  (i64)Lc * Lq, 0, 1, Lq,
       Tb,   (i64)Lc * Lc, 0, Lc, 1,
       nullptr, nullptr, 1.f, 0);
  // Bt = T @ c
  gemm(st, Lc, D, Lc, B, 1,
       Tb,   (i64)Lc * Lc, 0, Lc, 1,
       Cenc, (i64)Lc * D, 0, D, 1,
       Btb,  (i64)Lc * D, 0, D, 1,
       nullptr, nullptr, 1.f, 0);
  k_cqcat<<<EW((i64)B * Lc * D)>>>(Cenc, Aatt, Btb, xcat, B, Lc, D);
  // resizer: 4D -> D
  gemm(st, B * Lc, D, 4 * D, 1, 1,
       xcat, 0, 0, 4 * D, 1,
       resizer_w, 0, 0, 1, 4 * D,
       X0, 0, 0, D, 1,
       resizer_b, nullptr, 1.f, 0);

  // ----- model encoder stacks -----
  auto stack = [&](const float* in, float* out) {
    const float* curp = in;
    for (int i = 0; i < 7; i++) {
      float* dst = (i == 6) ? out : ((i & 1) ? zB : zA);
      run_block(st, sc, m0[i], curp, dst, B, Lc, D, D);
      curp = dst;
    }
  };
  stack(X0, M1);
  stack(M1, M2);
  stack(M2, M3);

  // ----- output heads -----
  const int KH = Lc * 2 * D;   // 102400
  float* out_f = (float*)d_out;
  k_catMM<<<EW((i64)B * Lc * D)>>>(M1, M2, hcat, B, Lc, D);
  gemm(st, B, Lc, KH, 1, 1,
       hcat, 0, 0, (i64)KH, 1,
       start_w, 0, 0, 1, (i64)KH,
       htmp, 0, 0, Lc, 1,
       start_b, nullptr, 1.f, 1);
  k_logsoftmax<<<dim3(B), dim3(256), 0, st>>>(htmp, out_f, Lc);

  k_catMM<<<EW((i64)B * Lc * D)>>>(M1, M3, hcat, B, Lc, D);
  gemm(st, B, Lc, KH, 1, 1,
       hcat, 0, 0, (i64)KH, 1,
       end_w, 0, 0, 1, (i64)KH,
       htmp, 0, 0, Lc, 1,
       end_b, nullptr, 1.f, 1);
  k_logsoftmax<<<dim3(B), dim3(256), 0, st>>>(htmp, out_f + (i64)B * Lc, Lc);
}
